// MultiHeadAttention2d_5480378270115
// MI455X (gfx1250) — matrix operand order repaired and checked by reading, not compiled, not run on hardware
//
#include <hip/hip_runtime.h>

// ---------------------------------------------------------------------------
// MultiHeadAttention2d for MI455X (gfx1250): bf16 WMMA everywhere.
// bs=16, C=512, H=W=32 (N=1024), heads=8, dim=64.
// ---------------------------------------------------------------------------

typedef __attribute__((ext_vector_type(16))) __bf16 v16bf;
typedef __attribute__((ext_vector_type(8)))  __bf16 bf16x8;
typedef __attribute__((ext_vector_type(8)))  float  v8f;

#define BSZ   16
#define CCH   512
#define NPIX  1024
#define NHEAD 8
#define DIMH  64
#define OCH   1536   // 3*C

static __device__ __forceinline__ __bf16 f2bf(float f) {
  union { float f; unsigned u; } a; a.f = f;
  unsigned r = a.u + 0x7FFFu + ((a.u >> 16) & 1u);   // round-to-nearest-even
  union { unsigned short s; __bf16 b; } o;
  o.s = (unsigned short)(r >> 16);
  return o.b;
}

// Build a 16-element bf16 fragment from two 8-element (16B) chunks.
static __device__ __forceinline__ v16bf ld16(const __bf16* lo, const __bf16* hi) {
  bf16x8 a = *(const bf16x8*)lo;
  bf16x8 b = *(const bf16x8*)hi;
  return __builtin_shufflevector(a, b, 0,1,2,3,4,5,6,7,8,9,10,11,12,13,14,15);
}
static __device__ __forceinline__ v16bf ld16c(const __bf16* p) { return ld16(p, p + 8); }

static __device__ __forceinline__ v8f vzero8() {
  v8f v;
#pragma unroll
  for (int i = 0; i < 8; ++i) v[i] = 0.0f;
  return v;
}

static __device__ __forceinline__ v8f wmma_bf16(v16bf a, v16bf b, v8f c) {
  return __builtin_amdgcn_wmma_f32_16x16x32_bf16(false, a, false, b, (short)0, c, false, false);
}

// ---------------------------------------------------------------------------
// Prep: xs = x + pos_enc, stored transposed+bf16 as X^T [b][n][c].
// 64x64 LDS tile transpose: coalesced float4 reads, 16B bf16 writes.
// ---------------------------------------------------------------------------
__global__ __launch_bounds__(256) void k_prep_x(const float* __restrict__ x,
                                                const float* __restrict__ pos,
                                                __bf16* __restrict__ xbf) {
  __shared__ __bf16 tile[64][64 + 8];   // +8 halfs (16B) pad; row stride 144B (16B mult.)
  const int b  = blockIdx.z;
  const int c0 = blockIdx.y * 64;
  const int n0 = blockIdx.x * 64;
  const int t  = threadIdx.x;
  {
    const int cl = t >> 2;            // 0..63: channel row
    const int nq = (t & 3) * 16;      // 16 pixels per thread
    const size_t base = (size_t)(b * CCH + c0 + cl) * NPIX + n0 + nq;
#pragma unroll
    for (int q = 0; q < 4; ++q) {
      float4 vx = *(const float4*)(x + base + q * 4);
      float4 vp = *(const float4*)(pos + base + q * 4);
      tile[nq + q * 4 + 0][cl] = f2bf(vx.x + vp.x);
      tile[nq + q * 4 + 1][cl] = f2bf(vx.y + vp.y);
      tile[nq + q * 4 + 2][cl] = f2bf(vx.z + vp.z);
      tile[nq + q * 4 + 3][cl] = f2bf(vx.w + vp.w);
    }
  }
  __syncthreads();
  {
    const int nl = t >> 2;            // pixel row of X^T
    const int ch = (t & 3) * 16;      // 16 channels -> 32B contiguous
    __bf16* pd = xbf + (size_t)(b * NPIX + n0 + nl) * CCH + c0 + ch;
    *(bf16x8*)(pd)     = *(const bf16x8*)&tile[nl][ch];
    *(bf16x8*)(pd + 8) = *(const bf16x8*)&tile[nl][ch + 8];
  }
}

__global__ __launch_bounds__(256) void k_prep_w(const float* __restrict__ W,
                                                __bf16* __restrict__ Wb) {
  int idx = blockIdx.x * 256 + threadIdx.x;          // [0, 1536*512)
  Wb[idx] = f2bf(W[idx]);
}

// ---------------------------------------------------------------------------
// QKV projection: P[o][n] = sum_c W[o][c] * X[c][n]  (+ bias), per batch.
// Block = 8 waves (2x4), block tile 128(o) x 256(n), wave tile 64x64.
// Epilogue (region uniform per block, from blockIdx.y):
//   o in [0,512)    -> QT[b][h][n][d] (scaled by 1/sqrt(dim))
//   o in [512,1024) -> KT[b][h][m][d]
//   o in [1024,1536)-> V [b][h][d][m]
// ---------------------------------------------------------------------------
__global__ __launch_bounds__(256) void k_qkv(const __bf16* __restrict__ Wb,
                                             const __bf16* __restrict__ Xb,   // [BSZ][NPIX][CCH]
                                             const float*  __restrict__ bias,
                                             __bf16* __restrict__ QT,
                                             __bf16* __restrict__ KT,
                                             __bf16* __restrict__ Vt) {
  const int lane = threadIdx.x & 31;
  const int wave = threadIdx.x >> 5;
  const int g    = lane >> 4;            // lane-group (0: lanes 0-15, 1: 16-31)
  const int l16  = lane & 15;
  const int b    = blockIdx.z;
  const int obase = blockIdx.y * 128 + (wave >> 2) * 64;
  const int nbase = blockIdx.x * 256 + (wave & 3) * 64;
  const __bf16* Xbb = Xb + (size_t)b * NPIX * CCH;

  v8f acc[4][4];
#pragma unroll
  for (int mi = 0; mi < 4; ++mi)
#pragma unroll
    for (int ni = 0; ni < 4; ++ni) acc[mi][ni] = vzero8();

  for (int kk = 0; kk < CCH; kk += 32) {
    v16bf af[4], bfr[4];
#pragma unroll
    for (int mi = 0; mi < 4; ++mi) {
      // A layout (16x32 bf16): g=0 -> K{0..7,16..23}, g=1 -> K{8..15,24..31}
      const __bf16* pA = Wb + (size_t)(obase + mi * 16 + l16) * CCH + kk + g * 8;
      af[mi] = ld16(pA, pA + 16);
    }
#pragma unroll
    for (int ni = 0; ni < 4; ++ni) {
      // B layout (32x16 bf16): col = l16, g=0 -> K{0..7,16..23}, g=1 -> K{8..15,24..31}
      const __bf16* pB = Xbb + (size_t)(nbase + ni * 16 + l16) * CCH + kk + g * 8;
      bfr[ni] = ld16(pB, pB + 16);
    }
#pragma unroll
    for (int mi = 0; mi < 4; ++mi)
#pragma unroll
      for (int ni = 0; ni < 4; ++ni)
        acc[mi][ni] = wmma_bf16(af[mi], bfr[ni], acc[mi][ni]);
  }

  // Uniform per block: blockIdx.y in [0,12), 4 y-blocks per 512-row region.
  const int region = (int)(blockIdx.y >> 2);   // 0=Q, 1=K, 2=V (scalar branch)
#pragma unroll
  for (int mi = 0; mi < 4; ++mi) {
#pragma unroll
    for (int r = 0; r < 8; ++r) {
      const int o  = obase + mi * 16 + r + g * 8;
      const float bo = bias[o];
      const int oo = o & 511;
      const int h  = oo >> 6;
      const int d  = oo & 63;
#pragma unroll
      for (int ni = 0; ni < 4; ++ni) {
        const int n = nbase + ni * 16 + l16;
        const float val = acc[mi][ni][r] + bo;
        if (region == 0) {
          QT[((size_t)((b * NHEAD + h) * NPIX + n) * DIMH) + d] = f2bf(val * 0.125f);
        } else if (region == 1) {
          KT[((size_t)((b * NHEAD + h) * NPIX + n) * DIMH) + d] = f2bf(val);
        } else {
          Vt[((size_t)((b * NHEAD + h) * DIMH + d) * NPIX) + n] = f2bf(val);
        }
      }
    }
  }
}

// ---------------------------------------------------------------------------
// Flash attention: each wave owns 16 query rows, streams K/V in m-chunks of 32
// with online softmax. O^T accumulated in f32, residual fused at the end.
// ---------------------------------------------------------------------------
__global__ __launch_bounds__(256) void k_attn(const __bf16* __restrict__ QT,
                                              const __bf16* __restrict__ KT,
                                              const __bf16* __restrict__ Vt,
                                              const float*  __restrict__ x,
                                              const float*  __restrict__ pos,
                                              float* __restrict__ out) {
  __shared__ __bf16 plds[8][16][32];      // per-wave P tile (D-layout -> A-layout relay)

  const int lane = threadIdx.x & 31;
  const int wave = threadIdx.x >> 5;
  const int g    = lane >> 4;
  const int l16  = lane & 15;
  const int bh   = blockIdx.x >> 3;                       // 0..127
  const int nbase = ((blockIdx.x & 7) * 8 + wave) * 16;   // 16-row query tile
  const int b = bh >> 3;
  const int h = bh & 7;

  const __bf16* Qb = QT + (size_t)bh * NPIX * DIMH;
  const __bf16* Kb = KT + (size_t)bh * NPIX * DIMH;
  const __bf16* Vb = Vt + (size_t)bh * DIMH * NPIX;

  // Q A-fragments (16 rows x 64 d), loaded once.
  v16bf qa[2];
#pragma unroll
  for (int kc = 0; kc < 2; ++kc) {
    const __bf16* p = Qb + (size_t)(nbase + l16) * DIMH + kc * 32 + g * 8;
    qa[kc] = ld16(p, p + 16);
  }

  v8f oacc[4];
#pragma unroll
  for (int j = 0; j < 4; ++j) oacc[j] = vzero8();
  float mrow[8], lrow[8];
#pragma unroll
  for (int r = 0; r < 8; ++r) { mrow[r] = -3.0e38f; lrow[r] = 0.0f; }

  for (int mb = 0; mb < NPIX; mb += 32) {
    // Prefetch next m-chunk of K (32 rows) and V (64 rows) into cache.
    if (mb + 32 < NPIX) {
      __builtin_prefetch(Kb + (size_t)(mb + 32 + lane) * DIMH, 0, 1);
      __builtin_prefetch(Vb + (size_t)lane * NPIX + (mb + 32), 0, 1);
      __builtin_prefetch(Vb + (size_t)(32 + lane) * NPIX + (mb + 32), 0, 1);
    }

    // K B-fragments: S tile is 16(n) x 32(m), K-dim d=64 -> 2 chunks.
    v16bf kfr[2][2];
#pragma unroll
    for (int ms = 0; ms < 2; ++ms)
#pragma unroll
      for (int kc = 0; kc < 2; ++kc) {
        const __bf16* p = Kb + (size_t)(mb + ms * 16 + l16) * DIMH + kc * 32 + g * 8;
        kfr[ms][kc] = ld16(p, p + 16);
      }
    v8f s0 = vzero8(), s1 = vzero8();
    s0 = wmma_bf16(qa[0], kfr[0][0], s0);
    s0 = wmma_bf16(qa[1], kfr[0][1], s0);
    s1 = wmma_bf16(qa[0], kfr[1][0], s1);
    s1 = wmma_bf16(qa[1], kfr[1][1], s1);

    // Online softmax: row r lives in acc element r across the 16 lanes of g.
#pragma unroll
    for (int r = 0; r < 8; ++r) {
      float rm = fmaxf(s0[r], s1[r]);
#pragma unroll
      for (int msk = 1; msk < 16; msk <<= 1) rm = fmaxf(rm, __shfl_xor(rm, msk, 32));
      const float mnew = fmaxf(mrow[r], rm);
      const float corr = __expf(mrow[r] - mnew);
      const float p0 = __expf(s0[r] - mnew);
      const float p1 = __expf(s1[r] - mnew);
      float rs = p0 + p1;
#pragma unroll
      for (int msk = 1; msk < 16; msk <<= 1) rs += __shfl_xor(rs, msk, 32);
      lrow[r] = lrow[r] * corr + rs;
      mrow[r] = mnew;
#pragma unroll
      for (int j = 0; j < 4; ++j) oacc[j][r] *= corr;
      plds[wave][r + g * 8][l16]      = f2bf(p0);
      plds[wave][r + g * 8][16 + l16] = f2bf(p1);
    }
    asm volatile("s_wait_dscnt 0" ::: "memory");   // wave-local LDS RAW fence

    // Re-read P in A layout (two ds_load_b128 per lane).
    v16bf pa = ld16(&plds[wave][l16][g * 8], &plds[wave][l16][16 + g * 8]);

    // V B-fragments: B[m(32)][d(16)] per 16-wide d tile.
    v16bf vfr[4];
#pragma unroll
    for (int j = 0; j < 4; ++j) {
      const __bf16* p = Vb + (size_t)(j * 16 + l16) * NPIX + mb + g * 8;
      vfr[j] = ld16(p, p + 16);
    }
#pragma unroll
    for (int j = 0; j < 4; ++j)
      oacc[j] = wmma_bf16(pa, vfr[j], oacc[j]);
  }

  // Epilogue: normalize, fuse residual x + pos_enc, write f32 output.
#pragma unroll
  for (int r = 0; r < 8; ++r) {
    const float inv = 1.0f / lrow[r];
    const int n = nbase + r + g * 8;
#pragma unroll
    for (int j = 0; j < 4; ++j) {
      const int c = h * DIMH + j * 16 + l16;
      const size_t idx = (size_t)(b * CCH + c) * NPIX + n;
      out[idx] = x[idx] + pos[idx] + oacc[j][r] * inv;
    }
  }
}

// ---------------------------------------------------------------------------
// Launcher. Workspace layout (bytes):
//   [0,        16777216) X^T  bf16 [b][n][c]
//   [16777216, 18350080) W    bf16 [o][c]
//   [18350080, 35127296) Q^T  bf16 [b][h][n][d]  (pre-scaled)
//   [35127296, 51904512) K^T  bf16 [b][h][m][d]
//   [51904512, 68681728) V    bf16 [b][h][d][m]
// ---------------------------------------------------------------------------
extern "C" void kernel_launch(void* const* d_in, const int* in_sizes, int n_in,
                              void* d_out, int out_size, void* d_ws, size_t ws_size,
                              hipStream_t stream) {
  const float* x    = (const float*)d_in[0];
  const float* pos  = (const float*)d_in[1];
  const float* W    = (const float*)d_in[2];
  const float* bias = (const float*)d_in[3];
  float* out = (float*)d_out;

  char* ws = (char*)d_ws;
  __bf16* xbf = (__bf16*)(ws);
  __bf16* Wb  = (__bf16*)(ws + 16777216);
  __bf16* QT  = (__bf16*)(ws + 18350080);
  __bf16* KT  = (__bf16*)(ws + 35127296);
  __bf16* Vt  = (__bf16*)(ws + 51904512);

  dim3 gprep(NPIX / 64, CCH / 64, BSZ);    // (16, 8, 16)
  k_prep_x<<<gprep, 256, 0, stream>>>(x, pos, xbf);
  k_prep_w<<<(OCH * CCH) / 256, 256, 0, stream>>>(W, Wb);

  dim3 gqkv(NPIX / 256, OCH / 128, BSZ);   // (4, 12, 16)
  k_qkv<<<gqkv, 256, 0, stream>>>(Wb, xbf, bias, QT, KT, Vt);

  k_attn<<<BSZ * NHEAD * (NPIX / 16 / 8), 256, 0, stream>>>(QT, KT, Vt, x, pos, out);
}
